// Attention_137438953963
// MI455X (gfx1250) — matrix operand order repaired and checked by reading, not compiled, not run on hardware
//
#include <hip/hip_runtime.h>
#include <hip/hip_bf16.h>

#define B_ 16
#define T_ 4096
#define D_ 1024
#define U_ 1024

typedef __attribute__((ext_vector_type(16))) __bf16 v16bf;
typedef __attribute__((ext_vector_type(8)))  float  v8f;

// Types for the gfx1250 async load-to-LDS builtin:
//   void __builtin_amdgcn_global_load_async_to_lds_b128(v4i as1*, v4i as3*, Ii, Ii)
typedef int v4i __attribute__((vector_size(16)));
typedef __attribute__((address_space(1))) v4i gv4i;
typedef __attribute__((address_space(3))) v4i lv4i;

// fp32 -> bf16, round-to-nearest-even
__device__ __forceinline__ __bf16 f2bf(float x) {
    unsigned u = __builtin_bit_cast(unsigned, x);
    u += 0x7FFFu + ((u >> 16) & 1u);
    return __builtin_bit_cast(__bf16, (unsigned short)(u >> 16));
}

// ---------------------------------------------------------------------------
// Kernel 1: hb[b,u] = hidden[b,:] @ W1[:,u] + b1[u] + b2[u]   (tiny)
// ---------------------------------------------------------------------------
__global__ void hproj_kernel(const float* __restrict__ h,
                             const float* __restrict__ W1,
                             const float* __restrict__ b1,
                             const float* __restrict__ b2,
                             float* __restrict__ hb) {
    int idx = blockIdx.x * 256 + threadIdx.x;   // 0..16383
    int b = idx >> 10, u = idx & (U_ - 1);
    const float* hr = h + b * D_;
    float acc = 0.f;
#pragma unroll 4
    for (int d = 0; d < D_; ++d) acc += hr[d] * W1[(size_t)d * U_ + u];
    hb[idx] = acc + b1[u] + b2[u];
}

// ---------------------------------------------------------------------------
// Kernel 2: W2 -> bf16, pre-swizzled into WMMA B-fragment layout.
//  fidx = s*2048 + ntile*32 + lane; elem e carries K = s*32 + 16*(lane/16) + e,
//  column n = ntile*16 + lane%16.
// ---------------------------------------------------------------------------
__global__ void w2_swizzle_kernel(const float* __restrict__ W2,
                                  __bf16* __restrict__ w2bf) {
    int fidx = blockIdx.x * 256 + threadIdx.x;  // 0..65535
    int l  = fidx & 31;
    int nt = (fidx >> 5) & 63;
    int s  = fidx >> 11;
    int n  = nt * 16 + (l & 15);
    int kb = s * 32 + 16 * (l >> 4);
    v16bf f;
#pragma unroll
    for (int e = 0; e < 16; ++e) f[e] = f2bf(W2[(size_t)(kb + e) * U_ + n]);
    ((v16bf*)w2bf)[fidx] = f;
}

// ---------------------------------------------------------------------------
// Kernel 3: enc -> bf16, pre-swizzled into WMMA A-fragment layout.
//  Tile = (b, tt): 16 rows. fidx = ((b*256+tt)*32 + s)*32 + l.
//  lane l: m=l%16, hf=l/16; elems 0..15 = row[s*32+16hf+0..15].
// ---------------------------------------------------------------------------
__global__ void enc_swizzle_kernel(const float* __restrict__ enc,
                                   __bf16* __restrict__ encbf) {
    int fidx = blockIdx.x * 256 + threadIdx.x;  // 0 .. 4M-1
    int l  = fidx & 31;
    int s  = (fidx >> 5) & 31;
    int tt = (fidx >> 10) & 255;
    int b  = fidx >> 18;
    int m = l & 15, hf = l >> 4;
    const float* row = enc + ((size_t)b * T_ + tt * 16 + m) * D_ + s * 32 + 16 * hf;
    float4 a0 = *(const float4*)(row);
    float4 a1 = *(const float4*)(row + 4);
    float4 a2 = *(const float4*)(row + 8);
    float4 a3 = *(const float4*)(row + 12);
    v16bf f;
    f[0]  = f2bf(a0.x); f[1]  = f2bf(a0.y); f[2]  = f2bf(a0.z); f[3]  = f2bf(a0.w);
    f[4]  = f2bf(a1.x); f[5]  = f2bf(a1.y); f[6]  = f2bf(a1.z); f[7]  = f2bf(a1.w);
    f[8]  = f2bf(a2.x); f[9]  = f2bf(a2.y); f[10] = f2bf(a2.z); f[11] = f2bf(a2.w);
    f[12] = f2bf(a3.x); f[13] = f2bf(a3.y); f[14] = f2bf(a3.z); f[15] = f2bf(a3.w);
    ((v16bf*)encbf)[fidx] = f;
}

// ---------------------------------------------------------------------------
// Kernel 4: fused score GEMM on pre-swizzled bf16 operands.
//  Per WG: 16 rows (one b, 16 t). 32 KB of A fragments moved to LDS with
//  GLOBAL_LOAD_ASYNC_TO_LDS_B128 (ASYNCcnt) when available, else a sync copy.
//  Waves split N=1024 (8 tiles of 16 each); inner loop:
//  1 LDS A-frag + 8x (32B B-frag + v_wmma_f32_16x16x32_bf16).
//  score[b,t] = sum_n tanh(acc + hb[b,n]) * V[n] + bv
// ---------------------------------------------------------------------------
__global__ void score_gemm_kernel(const __bf16* __restrict__ encbf,
                                  const __bf16* __restrict__ w2bf,
                                  const float* __restrict__ hb,
                                  const float* __restrict__ V,
                                  const float* __restrict__ bv,
                                  float* __restrict__ scores) {
    __shared__ __align__(32) __bf16 lA[32 * 32 * 16];   // 1024 fragments = 32 KB
    __shared__ float lds_score[16];

    const int tid  = threadIdx.x;
    const int wave = tid >> 5;
    const int lane = tid & 31;
    const int nlane = lane & 15;
    const int half  = lane >> 4;

    const int b     = blockIdx.x >> 8;
    const int trow0 = (blockIdx.x & 255) * 16;

    const v16bf* src = (const v16bf*)encbf + (size_t)blockIdx.x * 1024;
    if (tid < 16) lds_score[tid] = 0.f;

#if __has_builtin(__builtin_amdgcn_global_load_async_to_lds_b128)
    // Direct memory -> LDS copy of the contiguous 32 KB tile (no VGPR bounce).
    {
        const char* gsrc = (const char*)src;
        char*       ldst = (char*)lA;
#pragma unroll
        for (int i = 0; i < 8; ++i) {
            int idx16 = tid + i * 256;          // 16-byte chunk index, 2048 total
            __builtin_amdgcn_global_load_async_to_lds_b128(
                (gv4i*)(gsrc + (size_t)idx16 * 16),
                (lv4i*)(ldst + idx16 * 16), 0, 0);
        }
        asm volatile("s_wait_asynccnt 0x0" ::: "memory");
    }
#else
    {
        v16bf* dstv = (v16bf*)lA;
        for (int i = tid; i < 1024; i += 256) dstv[i] = src[i];
    }
#endif
    __syncthreads();

    const v16bf* lAv = (const v16bf*)lA;
    const v16bf* w2v = (const v16bf*)w2bf;

    v8f acc[8];
#pragma unroll
    for (int i = 0; i < 8; ++i) acc[i] = (v8f){0.f,0.f,0.f,0.f,0.f,0.f,0.f,0.f};

    for (int s = 0; s < 32; ++s) {
        v16bf afrag = lAv[s * 32 + lane];
        const v16bf* bp = w2v + (size_t)s * 2048 + wave * 256 + lane;
#pragma unroll
        for (int i = 0; i < 8; ++i) {
            acc[i] = __builtin_amdgcn_wmma_f32_16x16x32_bf16(
                false, afrag, false, bp[i * 32], (short)0, acc[i], false, false);
        }
    }

    // Epilogue: tanh + dot with V. C/D: vgpr r -> row (r + 8*half), col per lane.
    const int n0 = wave * 128 + nlane;
    float sc[8];
#pragma unroll
    for (int r = 0; r < 8; ++r) sc[r] = 0.f;
#pragma unroll
    for (int i = 0; i < 8; ++i) {
        const int n = n0 + i * 16;
        const float hbv = hb[b * U_ + n];
        const float vv  = V[n];
#pragma unroll
        for (int r = 0; r < 8; ++r)
            sc[r] += tanhf(acc[i][r] + hbv) * vv;
    }
#pragma unroll
    for (int r = 0; r < 8; ++r) {
        float v = sc[r];
        v += __shfl_xor(v, 1, 16);
        v += __shfl_xor(v, 2, 16);
        v += __shfl_xor(v, 4, 16);
        v += __shfl_xor(v, 8, 16);
        if (nlane == 0) atomicAdd(&lds_score[half * 8 + r], v);
    }
    __syncthreads();
    if (tid < 16)
        scores[(size_t)b * T_ + trow0 + tid] = lds_score[tid] + bv[0];
}

// ---------------------------------------------------------------------------
// Kernel 5: softmax over T per batch (one WG per b)
// ---------------------------------------------------------------------------
__global__ void softmax_kernel(const float* __restrict__ scores,
                               float* __restrict__ wts) {
    const int b = blockIdx.x, tid = threadIdx.x;
    const int wave = tid >> 5, lane = tid & 31;
    __shared__ float red[8];
    const float* sb = scores + (size_t)b * T_;

    float vals[16], lmax = -3.4e38f;
#pragma unroll
    for (int i = 0; i < 16; ++i) {
        vals[i] = sb[i * 256 + tid];
        lmax = fmaxf(lmax, vals[i]);
    }
#pragma unroll
    for (int off = 16; off >= 1; off >>= 1)
        lmax = fmaxf(lmax, __shfl_xor(lmax, off, 32));
    if (lane == 0) red[wave] = lmax;
    __syncthreads();
    float gmax = red[0];
#pragma unroll
    for (int i = 1; i < 8; ++i) gmax = fmaxf(gmax, red[i]);

    float lsum = 0.f;
#pragma unroll
    for (int i = 0; i < 16; ++i) { vals[i] = __expf(vals[i] - gmax); lsum += vals[i]; }
#pragma unroll
    for (int off = 16; off >= 1; off >>= 1)
        lsum += __shfl_xor(lsum, off, 32);
    __syncthreads();
    if (lane == 0) red[wave] = lsum;
    __syncthreads();
    float gsum = 0.f;
#pragma unroll
    for (int i = 0; i < 8; ++i) gsum += red[i];
    const float inv = 1.0f / gsum;
#pragma unroll
    for (int i = 0; i < 16; ++i)
        wts[(size_t)b * T_ + i * 256 + tid] = vals[i] * inv;
}

// ---------------------------------------------------------------------------
// Kernel 6: context[b,d] = sum_t w[b,t] * enc[b,t,d]  (bandwidth bound)
// ---------------------------------------------------------------------------
__global__ void context_kernel(const float* __restrict__ wts,
                               const float* __restrict__ enc,
                               float* __restrict__ ctx) {
    const int b = blockIdx.x, tc = blockIdx.y;   // tc in [0,8)
    const int tid = threadIdx.x;                 // owns 4 d's
    const float4* e4 = (const float4*)(enc + ((size_t)b * T_ + tc * 512) * D_);
    const float* wb = wts + (size_t)b * T_ + tc * 512;
    float4 acc = {0.f, 0.f, 0.f, 0.f};
    for (int t = 0; t < 512; ++t) {
        const float w = wb[t];
        float4 v = e4[(size_t)t * 256 + tid];
        acc.x += w * v.x; acc.y += w * v.y; acc.z += w * v.z; acc.w += w * v.w;
    }
    float* c = ctx + (size_t)b * D_ + tid * 4;
    atomicAdd(c + 0, acc.x);
    atomicAdd(c + 1, acc.y);
    atomicAdd(c + 2, acc.z);
    atomicAdd(c + 3, acc.w);
}

// ---------------------------------------------------------------------------
extern "C" void kernel_launch(void* const* d_in, const int* in_sizes, int n_in,
                              void* d_out, int out_size, void* d_ws, size_t ws_size,
                              hipStream_t stream) {
    const float* h   = (const float*)d_in[0];
    const float* enc = (const float*)d_in[1];
    const float* W1  = (const float*)d_in[2];
    const float* b1  = (const float*)d_in[3];
    const float* W2  = (const float*)d_in[4];
    const float* b2  = (const float*)d_in[5];
    const float* V   = (const float*)d_in[6];
    const float* bv  = (const float*)d_in[7];

    float* out = (float*)d_out;
    float* ctx = out;                    // [B, D]
    float* wts = out + B_ * D_;          // [B, T, 1]

    // Workspace layout (needs ~130.4 MB):
    float*  hb     = (float*)d_ws;                       //  64 KB
    float*  scores = hb + B_ * U_;                       // 256 KB
    __bf16* w2bf   = (__bf16*)(scores + B_ * T_);        //   2 MB
    __bf16* encbf  = w2bf + (size_t)D_ * U_;             // 128 MB

    (void)hipMemsetAsync(ctx, 0, (size_t)B_ * D_ * sizeof(float), stream);

    hproj_kernel<<<B_ * U_ / 256, 256, 0, stream>>>(h, W1, b1, b2, hb);
    w2_swizzle_kernel<<<D_ * U_ / 16 / 256, 256, 0, stream>>>(W2, w2bf);
    enc_swizzle_kernel<<<(int)((size_t)B_ * T_ * D_ / 16 / 256), 256, 0, stream>>>(enc, encbf);
    score_gemm_kernel<<<B_ * T_ / 16, 256, 0, stream>>>(encbf, w2bf, hb, V, bv, scores);
    softmax_kernel<<<B_, 256, 0, stream>>>(scores, wts);
    context_kernel<<<dim3(B_, 8), 256, 0, stream>>>(wts, enc, ctx);
}
